// TorchLMHeadCE_4887672783289
// MI455X (gfx1250) — compile-verified, matrix operand order checked by reading, not run on hardware
//
#include <hip/hip_runtime.h>
#include <hip/hip_bf16.h>

// ---------------------------------------------------------------------------
// Fused LM-head cross-entropy for MI455X (gfx1250, wave32, WMMA bf16)
//   logits = input @ weight^T  (never materialized; online softmax)
//   loss   = mean over valid rows of (logsumexp(logits[n]) - logits[n,target])
//
// Pass 0 (if ws fits): fp32 -> bf16 preconversion of input & weight into ws.
// Pass 1: streaming bf16 WMMA GEMM, double-buffered LDS filled with
//         GLOBAL_LOAD_ASYNC_TO_LDS_B128 (ASYNCcnt), online-softmax partials.
// Pass 2/3: split merge (LSE combine) and final mean.
// ---------------------------------------------------------------------------

typedef __attribute__((ext_vector_type(16))) __bf16 bf16x16;
typedef __attribute__((ext_vector_type(8)))  float  floatx8;
typedef int v4i_vs __attribute__((vector_size(16)));   // matches builtin param

#define MT   128            // rows per workgroup (8 waves x 16)
#define NVT  128            // V columns per chunk (8 sub-tiles of 16)
#define KC   128            // H (K) chunk staged in LDS
#define LSTR 136            // padded LDS row stride (bf16 elems): conflict-free
#define NEG_INF (-3.402823466e38f)
#define IGNORE_IDX (-100)

// gfx1250 async global->LDS copy (ASYNCcnt), with graceful fallback
#if defined(__HIP_DEVICE_COMPILE__) && __has_builtin(__builtin_amdgcn_global_load_async_to_lds_b128)
#define ASYNC_LDS 1
#else
#define ASYNC_LDS 0
#endif

__device__ __forceinline__ void copy16_g2l(const unsigned short* gsrc, __bf16* ldst) {
#if ASYNC_LDS
  __builtin_amdgcn_global_load_async_to_lds_b128(
      (__attribute__((address_space(1))) v4i_vs*)gsrc,
      (__attribute__((address_space(3))) v4i_vs*)ldst, 0, 0);
#else
  *(uint4*)ldst = *(const uint4*)gsrc;
#endif
}

__device__ __forceinline__ void wait_async0() {
#if ASYNC_LDS
#if __has_builtin(__builtin_amdgcn_s_wait_asynccnt)
  __builtin_amdgcn_s_wait_asynccnt(0);
#else
  asm volatile("s_wait_asynccnt 0x0" ::: "memory");
#endif
#endif
}

__device__ __forceinline__ unsigned pack_bf16(float a, float b) {
  unsigned ua = __float_as_uint(a);
  unsigned ub = __float_as_uint(b);
  ua += 0x7FFFu + ((ua >> 16) & 1u);   // round-to-nearest-even
  ub += 0x7FFFu + ((ub >> 16) & 1u);
  return (ua >> 16) | (ub & 0xFFFF0000u);
}

struct U32x8 { uint4 lo, hi; };

__device__ __forceinline__ bf16x16 load_frag(const __bf16* p) {
  U32x8 u;
  u.lo = ((const uint4*)p)[0];
  u.hi = ((const uint4*)p)[1];
  return __builtin_bit_cast(bf16x16, u);
}

// ---------------------------------------------------------------------------
// Pass 0: fp32 -> bf16 bulk conversion (vectorized, grid-stride)
// ---------------------------------------------------------------------------
__global__ __launch_bounds__(256)
void cvt_f32_bf16(const float* __restrict__ src, unsigned short* __restrict__ dst,
                  size_t n4 /* number of float4 groups */) {
  size_t i      = (size_t)blockIdx.x * blockDim.x + threadIdx.x;
  size_t stride = (size_t)gridDim.x * blockDim.x;
  for (; i < n4; i += stride) {
    float4 f = ((const float4*)src)[i];
    uint2 p; p.x = pack_bf16(f.x, f.y); p.y = pack_bf16(f.z, f.w);
    ((uint2*)dst)[i] = p;
  }
}

// ---------------------------------------------------------------------------
// Pass 1: streaming GEMM (bf16 WMMA) + online softmax partials per V-split
// PRE = operands preconverted to bf16 in workspace (async LDS fill path)
// ---------------------------------------------------------------------------
template <bool PRE>
__global__ __launch_bounds__(256)
void lmhead_ce_partial(const float* __restrict__ inF,
                       const float* __restrict__ wF,
                       const unsigned short* __restrict__ inB,
                       const unsigned short* __restrict__ wB,
                       const long long* __restrict__ target,
                       float* __restrict__ partM,
                       float* __restrict__ partS,
                       float* __restrict__ partT,
                       int N, int H, int V, int S)
{
  __shared__ __align__(16) __bf16 Alds[2][MT  * LSTR];
  __shared__ __align__(16) __bf16 Blds[2][NVT * LSTR];

  const int tid  = threadIdx.x;
  const int lane = tid & 31;
  const int wv   = tid >> 5;          // wave id 0..7 -> row tile
  const int ln15 = lane & 15;
  const int hf   = lane >> 4;         // 0: rows i (i=0..7), 1: rows i+8
  const int r0   = blockIdx.x * MT;
  const int sp   = blockIdx.y;
  const int vstart = (int)(((long long)sp       * V) / S);
  const int vend   = (int)(((long long)(sp + 1) * V) / S);
  const int nk     = H / KC;

  // Per-lane running softmax state for 8 rows (redundant across 16 N-lanes).
  float runM[8], runS[8], runT[8];
  int   tgt[8];
  #pragma unroll
  for (int i = 0; i < 8; ++i) {
    runM[i] = NEG_INF; runS[i] = 0.f; runT[i] = 0.f;
    int row = r0 + wv * 16 + i + hf * 8;
    tgt[i]  = (row < N) ? (int)target[row] : IGNORE_IDX;
  }

  // --------------------------- LDS fill helpers ---------------------------
  auto fill = [&](int buf, int kc0, int v0) {
    if constexpr (PRE) {
      // bf16 copy, 16B per lane-op; async on gfx1250 (ASYNCcnt)
      #pragma unroll
      for (int rep = 0; rep < (MT * KC / 8) / 256; ++rep) {   // 8 reps
        int idx = rep * 256 + tid;        // 16-byte chunk index
        int row = idx >> 4;               // KC/8 == 16 chunks per row
        int c8  = idx & 15;
        int grow = (r0 + row < N) ? (r0 + row) : (N - 1);
        copy16_g2l(inB + (size_t)grow * H + kc0 + c8 * 8,
                   &Alds[buf][row * LSTR + c8 * 8]);
      }
      #pragma unroll
      for (int rep = 0; rep < (NVT * KC / 8) / 256; ++rep) {
        int idx = rep * 256 + tid;
        int vr  = idx >> 4;
        int c8  = idx & 15;
        int v   = v0 + vr;
        if (v < V) {
          copy16_g2l(wB + (size_t)v * H + kc0 + c8 * 8,
                     &Blds[buf][vr * LSTR + c8 * 8]);
        } else {
          uint4 z = make_uint4(0u, 0u, 0u, 0u);
          *(uint4*)(&Blds[buf][vr * LSTR + c8 * 8]) = z;
        }
      }
    } else {
      // fp32 load + convert + ds_store fallback
      #pragma unroll
      for (int rep = 0; rep < (MT * KC / 4) / 256; ++rep) {   // 16 reps
        int idx = rep * 256 + tid;        // float4 index
        int row = idx >> 5;               // KC/4 == 32 float4 per row
        int c4  = idx & 31;
        int grow = (r0 + row < N) ? (r0 + row) : (N - 1);
        const float4 f = *(const float4*)(inF + (size_t)grow * H + kc0 + c4 * 4);
        uint2 p; p.x = pack_bf16(f.x, f.y); p.y = pack_bf16(f.z, f.w);
        *(uint2*)(&Alds[buf][row * LSTR + c4 * 4]) = p;
      }
      #pragma unroll
      for (int rep = 0; rep < (NVT * KC / 4) / 256; ++rep) {
        int idx = rep * 256 + tid;
        int vr  = idx >> 5;
        int c4  = idx & 31;
        int v   = v0 + vr;
        float4 f = make_float4(0.f, 0.f, 0.f, 0.f);
        if (v < V) {
          f = *(const float4*)(wF + (size_t)v * H + kc0 + c4 * 4);
          if (c4 == 0 && kc0 + KC < H)
            __builtin_prefetch(wF + (size_t)v * H + kc0 + KC, 0, 1);
        }
        uint2 p; p.x = pack_bf16(f.x, f.y); p.y = pack_bf16(f.z, f.w);
        *(uint2*)(&Blds[buf][vr * LSTR + c4 * 4]) = p;
      }
    }
  };

  // ------------------------------ main loop -------------------------------
  for (int v0 = vstart; v0 < vend; v0 += NVT) {
    floatx8 acc[8];
    #pragma unroll
    for (int nt = 0; nt < 8; ++nt) { floatx8 z = {}; acc[nt] = z; }

    fill(0, 0, v0);                       // prologue: chunk 0 -> buf 0

    for (int k = 0; k < nk; ++k) {
      const int buf = k & 1;
      if constexpr (PRE) wait_async0();   // our async fills for chunk k done
      __syncthreads();                    // all waves' fills visible; prev compute done
      if (k + 1 < nk) fill((k + 1) & 1, (k + 1) * KC, v0);  // overlap next copy

      const __bf16* Ab = Alds[buf];
      const __bf16* Bb = Blds[buf];
      #pragma unroll
      for (int kk = 0; kk < KC / 32; ++kk) {
        const int kb = kk * 32 + hf * 16;    // A/B frag: 16 contiguous bf16
        const bf16x16 afrag = load_frag(&Ab[(wv * 16 + ln15) * LSTR + kb]);
        #pragma unroll
        for (int nt = 0; nt < 8; ++nt) {
          const bf16x16 bfrag = load_frag(&Bb[(nt * 16 + ln15) * LSTR + kb]);
          acc[nt] = __builtin_amdgcn_wmma_f32_16x16x32_bf16(
              false, afrag, false, bfrag, (short)0, acc[nt], false, false);
        }
      }
    }
    __syncthreads();                      // protect buffers before next v-iter

    // ---- online softmax update over this 128-column chunk ----
    #pragma unroll
    for (int i = 0; i < 8; ++i) {
      float cm = NEG_INF;
      #pragma unroll
      for (int nt = 0; nt < 8; ++nt) {
        int col = v0 + nt * 16 + ln15;
        float val = (col < vend) ? acc[nt][i] : NEG_INF;
        cm = fmaxf(cm, val);
      }
      #pragma unroll
      for (int d = 1; d < 16; d <<= 1) cm = fmaxf(cm, __shfl_xor(cm, d, 32));

      float cs = 0.f;
      const int t = tgt[i];
      #pragma unroll
      for (int nt = 0; nt < 8; ++nt) {
        int col = v0 + nt * 16 + ln15;
        float val = (col < vend) ? acc[nt][i] : NEG_INF;
        cs += __expf(val - cm);
        if (col == t) runT[i] += acc[nt][i];   // target-logit capture
      }
      #pragma unroll
      for (int d = 1; d < 16; d <<= 1) cs += __shfl_xor(cs, d, 32);

      float nm = fmaxf(runM[i], cm);
      runS[i]  = runS[i] * __expf(runM[i] - nm) + cs * __expf(cm - nm);
      runM[i]  = nm;
    }
  }

  // ---- gather target-logit across the 16 N-lanes, write partials ----
  #pragma unroll
  for (int i = 0; i < 8; ++i) {
    #pragma unroll
    for (int d = 1; d < 16; d <<= 1) runT[i] += __shfl_xor(runT[i], d, 32);
  }
  if (ln15 == 0) {
    #pragma unroll
    for (int i = 0; i < 8; ++i) {
      int row = r0 + wv * 16 + i + hf * 8;
      if (row < N) {
        size_t off = (size_t)sp * N + row;
        partM[off] = runM[i];
        partS[off] = runS[i];
        partT[off] = runT[i];
      }
    }
  }
}

// ---------------------------------------------------------------------------
// Pass 2: merge V-splits per row (LSE combine), block-reduce nll & count
// ---------------------------------------------------------------------------
__global__ __launch_bounds__(256)
void lmhead_ce_merge(const float* __restrict__ partM,
                     const float* __restrict__ partS,
                     const float* __restrict__ partT,
                     const long long* __restrict__ target,
                     float* __restrict__ blockOut,
                     int N, int S, int nb)
{
  __shared__ float sm[256];
  __shared__ float sc[256];
  const int tid = threadIdx.x;
  const int r   = blockIdx.x * 256 + tid;

  float nll = 0.f, cnt = 0.f;
  if (r < N) {
    int t = (int)target[r];
    if (t != IGNORE_IDX) {
      float m = NEG_INF;
      for (int s = 0; s < S; ++s) m = fmaxf(m, partM[(size_t)s * N + r]);
      float sum = 0.f, tl = 0.f;
      for (int s = 0; s < S; ++s) {
        sum += partS[(size_t)s * N + r] * __expf(partM[(size_t)s * N + r] - m);
        tl  += partT[(size_t)s * N + r];
      }
      nll = (m + __logf(sum)) - tl;
      cnt = 1.f;
    }
  }
  sm[tid] = nll; sc[tid] = cnt;
  __syncthreads();
  for (int st = 128; st > 0; st >>= 1) {
    if (tid < st) { sm[tid] += sm[tid + st]; sc[tid] += sc[tid + st]; }
    __syncthreads();
  }
  if (tid == 0) { blockOut[blockIdx.x] = sm[0]; blockOut[nb + blockIdx.x] = sc[0]; }
}

// ---------------------------------------------------------------------------
// Pass 3: final scalar
// ---------------------------------------------------------------------------
__global__ void lmhead_ce_final(const float* __restrict__ blockOut,
                                float* __restrict__ out, int nb)
{
  float s = 0.f, c = 0.f;
  for (int i = threadIdx.x; i < nb; i += 32) { s += blockOut[i]; c += blockOut[nb + i]; }
  #pragma unroll
  for (int d = 16; d > 0; d >>= 1) { s += __shfl_xor(s, d, 32); c += __shfl_xor(c, d, 32); }
  if (threadIdx.x == 0) out[0] = s / fmaxf(c, 1.f);
}

// ---------------------------------------------------------------------------
extern "C" void kernel_launch(void* const* d_in, const int* in_sizes, int n_in,
                              void* d_out, int out_size, void* d_ws, size_t ws_size,
                              hipStream_t stream) {
  const float*      input  = (const float*)d_in[0];
  const float*      weight = (const float*)d_in[1];
  const long long*  target = (const long long*)d_in[2];

  const int N = in_sizes[2];
  const int H = in_sizes[0] / N;
  const int V = in_sizes[1] / H;

  const int nb = (N + 255) / 256;

  // partials first: 3*S*N + 2*nb floats
  int S = 8;
  while (S > 1 && 4 * (3 * (size_t)S * N + 2 * (size_t)nb) > ws_size) --S;
  size_t partBytes = 4 * (3 * (size_t)S * N + 2 * (size_t)nb);

  char*  wsBase   = (char*)d_ws;
  float* partM    = (float*)wsBase;
  float* partS    = partM + (size_t)S * N;
  float* partT    = partS + (size_t)S * N;
  float* blockOut = partT + (size_t)S * N;

  // bf16 preconversion buffers after partials (16B aligned)
  size_t off = (partBytes + 15) & ~(size_t)15;
  size_t bfBytes = 2 * ((size_t)N * H + (size_t)V * H);
  const bool pre = (off + bfBytes <= ws_size) && (H % KC == 0) && (H % 8 == 0);

  unsigned short* Abf = (unsigned short*)(wsBase + off);
  unsigned short* Wbf = Abf + (size_t)N * H;

  dim3 grid1((N + MT - 1) / MT, S);
  if (pre) {
    size_t a4 = ((size_t)N * H) / 4;
    size_t w4 = ((size_t)V * H) / 4;
    cvt_f32_bf16<<<2048, 256, 0, stream>>>(input,  Abf, a4);
    cvt_f32_bf16<<<4096, 256, 0, stream>>>(weight, Wbf, w4);
    lmhead_ce_partial<true><<<grid1, 256, 0, stream>>>(
        input, weight, Abf, Wbf, target, partM, partS, partT, N, H, V, S);
  } else {
    lmhead_ce_partial<false><<<grid1, 256, 0, stream>>>(
        input, weight, (const unsigned short*)nullptr, (const unsigned short*)nullptr,
        target, partM, partS, partT, N, H, V, S);
  }
  lmhead_ce_merge<<<nb, 256, 0, stream>>>(partM, partS, partT, target,
                                          blockOut, N, S, nb);
  lmhead_ce_final<<<1, 32, 0, stream>>>(blockOut, (float*)d_out, nb);
}
